// Mamba2Block_44487271252591
// MI455X (gfx1250) — matrix operand order repaired and checked by reading, not compiled, not run on hardware
//
#include <hip/hip_runtime.h>
#include <hip/hip_bf16.h>
#include <cstdint>
#include <cstddef>

// ---------------- problem constants (Mamba2 block) ----------------
#define D_MODEL 2048
#define D_INNER 4096
#define NHEADS  64
#define HEADDIM 64
#define D_STATE 128
#define CHUNK   64
#define NCHUNK  64              // 4096 / 64 per batch
#define BATCH   2
#define SEQLEN  4096
#define NTOK    (BATCH*SEQLEN)  // 8192 tokens
#define XPD     4416            // D_INNER + NHEADS + 2*D_STATE

#define USE_ASYNC 1

typedef __attribute__((ext_vector_type(16))) __bf16 v16bf;
typedef __attribute__((ext_vector_type(8)))  __bf16 v8bf;
typedef __attribute__((ext_vector_type(8)))  float  v8f;

// ---------------- bf16 helpers (raw ushort storage) ----------------
__device__ __forceinline__ unsigned short f2bf(float f) {
  unsigned u = __float_as_uint(f);
  return (unsigned short)((u + 0x7FFFu + ((u >> 16) & 1u)) >> 16);   // RNE
}
__device__ __forceinline__ float bf2f(unsigned short h) {
  return __uint_as_float(((unsigned)h) << 16);
}

// ---------------- CDNA5 async global->LDS copy (ASYNCcnt path) ----------------
__device__ __forceinline__ void async_b128(const unsigned short* g, unsigned short* l) {
#if USE_ASYNC
  unsigned ldsaddr = (unsigned)(uintptr_t)l;    // low 32 bits = LDS offset
  asm volatile("global_load_async_to_lds_b128 %0, %1, off"
               :: "v"(ldsaddr), "v"(g) : "memory");
#else
  *(uint4*)l = *(const uint4*)g;
#endif
}
__device__ __forceinline__ void wait_async0() {
#if USE_ASYNC
#if __has_builtin(__builtin_amdgcn_s_wait_asynccnt)
  __builtin_amdgcn_s_wait_asynccnt(0);
#else
  asm volatile("s_wait_asynccnt 0" ::: "memory");
#endif
#endif
}

// ---------------- WMMA helpers ----------------
__device__ __forceinline__ v8f wmma_bf16(v16bf a, v16bf b, v8f c) {
  return __builtin_amdgcn_wmma_f32_16x16x32_bf16(false, a, false, b, (short)0, c,
                                                 false, false);
}
// A operand (16x32, M x K). Lane l: M = l&15, half = l>>4.
__device__ __forceinline__ v16bf fragA(const unsigned short* row, int ks, int half) {
  v8bf lo = *(const v8bf*)(row + ks + 8*half);
  v8bf hi = *(const v8bf*)(row + ks + 16 + 8*half);
  return __builtin_shufflevector(lo, hi, 0,1,2,3,4,5,6,7,8,9,10,11,12,13,14,15);
}
// B operand (32x16, K x N), LDS stored as [N][K]. Lane l: N = l&15, half = l>>4.
__device__ __forceinline__ v16bf fragB(const unsigned short* row, int ks, int half) {
  v8bf lo = *(const v8bf*)(row + ks + 8*half);
  v8bf hi = *(const v8bf*)(row + ks + 16 + 8*half);
  return __builtin_shufflevector(lo, hi, 0,1,2,3,4,5,6,7,8,9,10,11,12,13,14,15);
}

// ---------------- fp32 -> bf16 convert ----------------
__global__ void cvt_bf16_kernel(const float* __restrict__ src,
                                unsigned short* __restrict__ dst, size_t n) {
  size_t i = (size_t)blockIdx.x * 256 + threadIdx.x;
  size_t stride = (size_t)gridDim.x * 256;
  for (; i < n; i += stride) dst[i] = f2bf(src[i]);
}

// ---------------- generic bf16 WMMA GEMM: C[M,N] = A[M,K] * W[N,K]^T ----------------
// Double-buffered LDS, async global->LDS copies pipelined against WMMA.
#define GBM 128
#define GBN 64
#define GBK 64
#define GLDA 72   // 144B rows: 16B aligned, conflict-free fragment reads

__global__ __launch_bounds__(256) void gemm_bf16_kernel(
    const unsigned short* __restrict__ A, const unsigned short* __restrict__ W,
    float* __restrict__ Cf, unsigned short* __restrict__ Cb,
    int M, int N, int K, int out_bf16)
{
  __shared__ __align__(16) unsigned short As[2][GBM * GLDA];
  __shared__ __align__(16) unsigned short Bs[2][GBN * GLDA];
  const int tid  = threadIdx.x;
  const int wave = tid >> 5, lane = tid & 31;
  const int half = lane >> 4, lr = lane & 15;
  const int m0 = blockIdx.x * GBM;
  const int n0 = blockIdx.y * GBN;
  const int wm = wave >> 1;     // 0..3 : 32-row slab
  const int wn = wave & 1;      // 0..1 : 32-col slab
  const int ar = tid >> 3, acg = (tid & 7) * 8;   // per-thread copy slot

  const v8f vzero = {0.f,0.f,0.f,0.f,0.f,0.f,0.f,0.f};
  v8f acc[2][2];
  acc[0][0]=vzero; acc[0][1]=vzero; acc[1][0]=vzero; acc[1][1]=vzero;

  auto issue = [&](int kt, int buf) {
#pragma unroll
    for (int i = 0; i < 4; ++i) {                  // A tile: 128x64 bf16
      int r = ar + i * 32;
      async_b128(A + (size_t)(m0 + r) * K + kt + acg, &As[buf][r * GLDA + acg]);
    }
#pragma unroll
    for (int i = 0; i < 2; ++i) {                  // W tile: 64x64 bf16
      int r = ar + i * 32;
      async_b128(W + (size_t)(n0 + r) * K + kt + acg, &Bs[buf][r * GLDA + acg]);
    }
  };

  issue(0, 0);
  wait_async0();
  __syncthreads();
  int cur = 0;
  for (int kt = 0; kt < K; kt += GBK) {
    if (kt + GBK < K) issue(kt + GBK, cur ^ 1);    // overlap copy with compute
    const unsigned short* Asb = &As[cur][0];
    const unsigned short* Bsb = &Bs[cur][0];
#pragma unroll
    for (int ks = 0; ks < GBK; ks += 32) {
      v16bf a0 = fragA(&Asb[(wm * 32 +  0 + lr) * GLDA], ks, half);
      v16bf a1 = fragA(&Asb[(wm * 32 + 16 + lr) * GLDA], ks, half);
      v16bf b0 = fragB(&Bsb[(wn * 32 +  0 + lr) * GLDA], ks, half);
      v16bf b1 = fragB(&Bsb[(wn * 32 + 16 + lr) * GLDA], ks, half);
      acc[0][0] = wmma_bf16(a0, b0, acc[0][0]);
      acc[0][1] = wmma_bf16(a0, b1, acc[0][1]);
      acc[1][0] = wmma_bf16(a1, b0, acc[1][0]);
      acc[1][1] = wmma_bf16(a1, b1, acc[1][1]);
    }
    wait_async0();
    __syncthreads();
    cur ^= 1;
  }
#pragma unroll
  for (int ti = 0; ti < 2; ++ti)
#pragma unroll
    for (int tj = 0; tj < 2; ++tj)
#pragma unroll
      for (int i = 0; i < 8; ++i) {
        int row = m0 + wm * 32 + ti * 16 + half * 8 + i;
        int col = n0 + wn * 32 + tj * 16 + lr;
        size_t off = (size_t)row * N + col;
        float v = acc[ti][tj][i];
        if (out_bf16) Cb[off] = f2bf(v); else Cf[off] = v;
      }
}

// ---------------- causal depthwise conv(4) + bias + silu ----------------
// 4 tokens per thread: 7 loads -> 4 outputs (groups never cross batch edge).
__global__ __launch_bounds__(256) void conv_silu_kernel(
    const unsigned short* __restrict__ xzb, const float* __restrict__ convw,
    const float* __restrict__ convb, unsigned short* __restrict__ out)
{
  int id = blockIdx.x * 256 + threadIdx.x;     // (NTOK/4) * D_INNER threads
  int ch = id & (D_INNER - 1);
  int g  = id >> 12;                           // token group (4 tokens)
  int token0 = g * 4;
  int l0 = token0 & (SEQLEN - 1);
  int bstart = token0 - l0;
  float w0 = convw[ch*4+0], w1 = convw[ch*4+1], w2 = convw[ch*4+2], w3 = convw[ch*4+3];
  float xv[7];
#pragma unroll
  for (int j = 0; j < 7; ++j) {
    int li = l0 - 3 + j;
    xv[j] = (li >= 0) ? bf2f(xzb[(size_t)(bstart + li) * 8192 + ch]) : 0.f;
  }
  float bias = convb[ch];
#pragma unroll
  for (int j = 0; j < 4; ++j) {
    float acc = bias + w0*xv[j] + w1*xv[j+1] + w2*xv[j+2] + w3*xv[j+3];
    float s = acc / (1.f + expf(-acc));        // silu
    out[(size_t)(token0 + j) * D_INNER + ch] = f2bf(s);
  }
}

// ---------------- dt head-mix + softplus ----------------
__global__ __launch_bounds__(64) void dt_kernel(
    const unsigned short* __restrict__ projb, const float* __restrict__ dtw,
    const float* __restrict__ dtb, float* __restrict__ dtbuf)
{
  __shared__ float raw[64];
  int token = blockIdx.x;
  int h = threadIdx.x;
  raw[h] = bf2f(projb[(size_t)token * XPD + D_INNER + h]);
  __syncthreads();
  float acc = dtb[h];
#pragma unroll 8
  for (int j = 0; j < 64; ++j) acc += dtw[h * 64 + j] * raw[j];
  float sp = (acc > 20.f) ? acc : log1pf(expf(acc));
  dtbuf[(size_t)token * 64 + h] = sp;
}

// ---------------- S1: per (b,chunk,head): cumsum, Y_diag, local states ----------------
// wide arrays stride 136 (272B rows), narrow stride 72 (144B rows)
__global__ __launch_bounds__(256) void chunk_kernel(
    const unsigned short* __restrict__ projb, const float* __restrict__ dtbuf,
    const float* __restrict__ A_log,
    float* __restrict__ Yd, unsigned short* __restrict__ states,
    float* __restrict__ Acs, float* __restrict__ chunkA)
{
  __shared__ __align__(16) unsigned char smem[54016];
  unsigned short* Cl  = (unsigned short*)(smem);           // [64][136]  phase A
  unsigned short* Bl  = (unsigned short*)(smem + 17408);   // [64][136]  phase A
  unsigned short* Xt  = (unsigned short*)(smem + 34816);   // [64][72]  Xs^T [p][t]
  unsigned short* XdT = (unsigned short*)(smem + 44032);   // [64][72]  (Xs*decay)^T
  unsigned short* Ml  = (unsigned short*)(smem);           // [64][72]  phase B (alias Cl)
  unsigned short* Bt  = (unsigned short*)(smem + 9216);    // [128][72] phase B (alias)
  float* dts  = (float*)(smem + 53248);
  float* acss = (float*)(smem + 53504);
  float* decs = (float*)(smem + 53760);

  const int tid = threadIdx.x;
  const int bc  = blockIdx.x;            // b*64 + chunk
  const int h   = blockIdx.y;
  const size_t tok0 = (size_t)bc * CHUNK;
  const int wave = tid >> 5, lane = tid & 31, half = lane >> 4, lr = lane & 15;
  const v8f vzero = {0.f,0.f,0.f,0.f,0.f,0.f,0.f,0.f};

  // B (64x128) as [s][n], C (64x128) as [t][n]: async, transform-free
#pragma unroll
  for (int i = 0; i < 4; ++i) {
    int id = tid + i * 256;
    int t = id >> 4, cg = (id & 15) * 8;
    async_b128(projb + (tok0 + t) * XPD + 4160 + cg, &Bl[t * 136 + cg]);
    async_b128(projb + (tok0 + t) * XPD + 4288 + cg, &Cl[t * 136 + cg]);
  }

  if (tid < 64) dts[tid] = dtbuf[(tok0 + tid) * 64 + h];
  __syncthreads();
  if (tid == 0) {
    float Ah = -expf(A_log[h]);
    float cs = 0.f;
    for (int t = 0; t < 64; ++t) { cs += Ah * dts[t]; acss[t] = cs; }
  }
  __syncthreads();
  if (tid < 64) {
    decs[tid] = expf(acss[63] - acss[tid]);
    Acs[(tok0 + tid) * 64 + h] = acss[tid];
    if (tid == 0) chunkA[(size_t)bc * 64 + h] = acss[63];
  }
  __syncthreads();
  // X (64 x 64): Xs = X*dt[t]; XdT additionally * decay[t]; stored transposed [p][t]
#pragma unroll
  for (int i = 0; i < 16; ++i) {
    int id = tid + i * 256;
    int t = id >> 6, p = id & 63;
    float x  = bf2f(projb[(tok0 + t) * XPD + h * 64 + p]);
    float xs = x * dts[t];
    Xt [p * 72 + t] = f2bf(xs);
    XdT[p * 72 + t] = f2bf(xs * decs[t]);
  }
  wait_async0();
  __syncthreads();

  // G = C * B^T  (M=t 64, N=s 64, K=n 128): 16 tiles, 2 per wave
  v8f gacc[2]; gacc[0] = vzero; gacc[1] = vzero;
#pragma unroll
  for (int q = 0; q < 2; ++q) {
    int tile = wave * 2 + q, ti = tile >> 2, tj = tile & 3;
#pragma unroll
    for (int ks = 0; ks < 128; ks += 32) {
      v16bf a = fragA(&Cl[(ti * 16 + lr) * 136], ks, half);
      v16bf b = fragB(&Bl[(tj * 16 + lr) * 136], ks, half);
      gacc[q] = wmma_bf16(a, b, gacc[q]);
    }
  }
  __syncthreads();   // all reads of Cl/Bl (as operands) done

  // Phase B: Ml[t][s] = G * L(t,s) into Cl's space; stage B^T via registers
  union { unsigned short u16[32]; uint4 v4[4]; } btmp;
  const int bn = tid & 127, tbase = (tid >> 7) * 32;
#pragma unroll
  for (int i = 0; i < 32; ++i) btmp.u16[i] = Bl[(tbase + i) * 136 + bn];  // in-LDS read
#pragma unroll
  for (int q = 0; q < 2; ++q) {
    int tile = wave * 2 + q, ti = tile >> 2, tj = tile & 3;
#pragma unroll
    for (int i = 0; i < 8; ++i) {
      int t = ti * 16 + half * 8 + i;
      int s = tj * 16 + lr;
      float L = (s <= t) ? expf(acss[t] - acss[s]) : 0.f;
      Ml[t * 72 + s] = f2bf(gacc[q][i] * L);
    }
  }
  __syncthreads();   // Bl reads complete everywhere before Bt overwrites it
#pragma unroll
  for (int i = 0; i < 4; ++i)
    *(uint4*)(&Bt[bn * 72 + tbase + i * 8]) = btmp.v4[i];   // ds_store_b128
  __syncthreads();

  // Y_diag = Ml @ Xs  (M=t, N=p, K=s=64): 16 tiles, 2 per wave
#pragma unroll
  for (int q = 0; q < 2; ++q) {
    int tile = wave * 2 + q, ti = tile >> 2, tj = tile & 3;
    v8f acc = vzero;
#pragma unroll
    for (int ks = 0; ks < 64; ks += 32) {
      v16bf a = fragA(&Ml[(ti * 16 + lr) * 72], ks, half);
      v16bf b = fragB(&Xt[(tj * 16 + lr) * 72], ks, half);
      acc = wmma_bf16(a, b, acc);
    }
#pragma unroll
    for (int i = 0; i < 8; ++i) {
      int t = ti * 16 + half * 8 + i;
      int p = tj * 16 + lr;
      Yd[(tok0 + t) * D_INNER + h * 64 + p] = acc[i];
    }
  }
  // states[p][n] = (Xs*decay)^T @ B  (M=p 64, N=n 128, K=t 64): 32 tiles, 4 per wave
  size_t sbase = ((size_t)bc * 64 + h) * 8192;
#pragma unroll
  for (int q = 0; q < 4; ++q) {
    int tile = wave * 4 + q, ti = tile >> 3, tj = tile & 7;
    v8f acc = vzero;
#pragma unroll
    for (int ks = 0; ks < 64; ks += 32) {
      v16bf a = fragA(&XdT[(ti * 16 + lr) * 72], ks, half);
      v16bf b = fragB(&Bt [(tj * 16 + lr) * 72], ks, half);
      acc = wmma_bf16(a, b, acc);
    }
#pragma unroll
    for (int i = 0; i < 8; ++i) {
      int p = ti * 16 + half * 8 + i;
      int n = tj * 16 + lr;
      states[sbase + p * 128 + n] = f2bf(acc[i]);
    }
  }
}

// ---------------- S2: sequential inter-chunk state scan (in-place) ----------------
__global__ __launch_bounds__(256) void scan_kernel(
    unsigned short* __restrict__ states, const float* __restrict__ chunkA)
{
  int b = blockIdx.x, h = blockIdx.y;
  int tid = threadIdx.x;
  float R[32];
#pragma unroll
  for (int i = 0; i < 32; ++i) R[i] = 0.f;
  for (int k = 0; k < NCHUNK; ++k) {
    size_t base = ((size_t)(b * 64 + k) * 64 + h) * 8192;
    float d = expf(chunkA[(size_t)(b * 64 + k) * 64 + h]);
#pragma unroll
    for (int i = 0; i < 32; ++i) {
      size_t off = base + (size_t)i * 256 + tid;
      float s = bf2f(states[off]);
      states[off] = f2bf(R[i]);        // emit states_in[k]
      R[i] = R[i] * d + s;             // carry
    }
  }
}

// ---------------- S3: Y_off + D-skip + silu(z) gating ----------------
__global__ __launch_bounds__(256) void out_gate_kernel(
    const unsigned short* __restrict__ projb, const unsigned short* __restrict__ states,
    const float* __restrict__ Yd, const float* __restrict__ Acs,
    const unsigned short* __restrict__ xzb, const float* __restrict__ Dv,
    unsigned short* __restrict__ Ygb)
{
  __shared__ __align__(16) unsigned short Csl[64 * 136];  // [t][n], pre-scaled exp(Acs)
  __shared__ __align__(16) unsigned short Sl [64 * 136];  // [p][n]
  __shared__ float exs[64];
  const int tid = threadIdx.x, bc = blockIdx.x, h = blockIdx.y;
  const size_t tok0 = (size_t)bc * CHUNK;
  const int wave = tid >> 5, lane = tid & 31, half = lane >> 4, lr = lane & 15;
  const v8f vzero = {0.f,0.f,0.f,0.f,0.f,0.f,0.f,0.f};

  size_t sbase = ((size_t)bc * 64 + h) * 8192;
#pragma unroll
  for (int i = 0; i < 4; ++i) {                  // states_in tile: async raw copy
    int id = tid + i * 256;
    int p = id >> 4, cg = (id & 15) * 8;
    async_b128(states + sbase + p * 128 + cg, &Sl[p * 136 + cg]);
  }
  if (tid < 64) exs[tid] = expf(Acs[(tok0 + tid) * 64 + h]);
  __syncthreads();
#pragma unroll
  for (int i = 0; i < 32; ++i) {
    int id = tid + i * 256;
    int t = id >> 7, n = id & 127;
    Csl[t * 136 + n] = f2bf(bf2f(projb[(tok0 + t) * XPD + 4288 + n]) * exs[t]);
  }
  wait_async0();
  __syncthreads();

  float Dh = Dv[h];
#pragma unroll
  for (int q = 0; q < 2; ++q) {      // Y_off (M=t, N=p, K=n=128): 16 tiles, 2/wave
    int tile = wave * 2 + q, ti = tile >> 2, tj = tile & 3;
    v8f acc = vzero;
#pragma unroll
    for (int ks = 0; ks < 128; ks += 32) {
      v16bf a = fragA(&Csl[(ti * 16 + lr) * 136], ks, half);
      v16bf b = fragB(&Sl [(tj * 16 + lr) * 136], ks, half);
      acc = wmma_bf16(a, b, acc);
    }
#pragma unroll
    for (int i = 0; i < 8; ++i) {
      int t = ti * 16 + half * 8 + i;
      int p = tj * 16 + lr;
      size_t token = tok0 + t;
      float y = acc[i] + Yd[token * D_INNER + h * 64 + p]
              + bf2f(projb[token * XPD + h * 64 + p]) * Dh;
      float z = bf2f(xzb[token * 8192 + 4096 + h * 64 + p]);
      y *= z / (1.f + expf(-z));
      Ygb[token * D_INNER + h * 64 + p] = f2bf(y);
    }
  }
}

// ---------------- host ----------------
extern "C" void kernel_launch(void* const* d_in, const int* in_sizes, int n_in,
                              void* d_out, int out_size, void* d_ws, size_t ws_size,
                              hipStream_t stream)
{
  (void)in_sizes; (void)n_in; (void)out_size; (void)ws_size;
  const float* x       = (const float*)d_in[0];
  const float* in_w    = (const float*)d_in[1];
  const float* conv_w  = (const float*)d_in[2];
  const float* conv_b  = (const float*)d_in[3];
  const float* xproj_w = (const float*)d_in[4];
  const float* dt_w    = (const float*)d_in[5];
  const float* dt_b    = (const float*)d_in[6];
  const float* A_log   = (const float*)d_in[7];
  const float* Dv      = (const float*)d_in[8];
  const float* out_w   = (const float*)d_in[9];
  float* out = (float*)d_out;

  char* ws = (char*)d_ws;
  size_t off = 0;
  auto alloc = [&](size_t bytes) -> char* {
    char* p = ws + off; off += (bytes + 255) & ~(size_t)255; return p;
  };
  unsigned short* xzb    = (unsigned short*)alloc((size_t)NTOK * 8192 * 2);        // 128MB
  unsigned short* projb  = (unsigned short*)alloc((size_t)NTOK * XPD * 2);         //  69MB
  unsigned short* xb     = (unsigned short*)alloc((size_t)NTOK * D_MODEL * 2);
  unsigned short* wb1    = (unsigned short*)alloc((size_t)2 * D_INNER * D_MODEL * 2);
  unsigned short* wb2    = (unsigned short*)alloc((size_t)XPD * D_INNER * 2);
  unsigned short* wb3    = (unsigned short*)alloc((size_t)D_MODEL * D_INNER * 2);
  unsigned short* xconvb = (unsigned short*)alloc((size_t)NTOK * D_INNER * 2);
  float*          dtbuf  = (float*)alloc((size_t)NTOK * NHEADS * 4);
  float*          Acsb   = (float*)alloc((size_t)NTOK * NHEADS * 4);
  float*          chkA   = (float*)alloc((size_t)BATCH * NCHUNK * NHEADS * 4);
  float*          Yd     = (float*)alloc((size_t)NTOK * D_INNER * 4);              // 128MB
  unsigned short* statesb= (unsigned short*)alloc((size_t)BATCH * NCHUNK * NHEADS * 8192 * 2); // 128MB
  unsigned short* Ygb    = (unsigned short*)alloc((size_t)NTOK * D_INNER * 2);

  cvt_bf16_kernel<<<4096, 256, 0, stream>>>(x, xb, (size_t)NTOK * D_MODEL);
  cvt_bf16_kernel<<<4096, 256, 0, stream>>>(in_w, wb1, (size_t)2 * D_INNER * D_MODEL);
  cvt_bf16_kernel<<<4096, 256, 0, stream>>>(xproj_w, wb2, (size_t)XPD * D_INNER);
  cvt_bf16_kernel<<<4096, 256, 0, stream>>>(out_w, wb3, (size_t)D_MODEL * D_INNER);

  // xz = x @ in_proj_w^T   (8192 x 8192 x 2048)
  gemm_bf16_kernel<<<dim3(NTOK / GBM, 8192 / GBN), 256, 0, stream>>>(
      xb, wb1, nullptr, xzb, NTOK, 8192, D_MODEL, 1);

  conv_silu_kernel<<<(NTOK / 4 * D_INNER) / 256, 256, 0, stream>>>(xzb, conv_w, conv_b, xconvb);

  // proj = silu(conv(x)) @ x_proj_w^T   (8192 x 4416 x 4096)
  gemm_bf16_kernel<<<dim3(NTOK / GBM, XPD / GBN), 256, 0, stream>>>(
      xconvb, wb2, nullptr, projb, NTOK, XPD, D_INNER, 1);

  dt_kernel<<<NTOK, 64, 0, stream>>>(projb, dt_w, dt_b, dtbuf);

  chunk_kernel<<<dim3(BATCH * NCHUNK, NHEADS), 256, 0, stream>>>(
      projb, dtbuf, A_log, Yd, statesb, Acsb, chkA);

  scan_kernel<<<dim3(BATCH, NHEADS), 256, 0, stream>>>(statesb, chkA);

  out_gate_kernel<<<dim3(BATCH * NCHUNK, NHEADS), 256, 0, stream>>>(
      projb, statesb, Yd, Acsb, xzb, Dv, Ygb);

  // out = Ygated @ out_proj_w^T   (8192 x 2048 x 4096)
  gemm_bf16_kernel<<<dim3(NTOK / GBM, D_MODEL / GBN), 256, 0, stream>>>(
      Ygb, wb3, out, nullptr, NTOK, D_MODEL, D_INNER, 0);
}
